// ArcFaceHead_82471962018284
// MI455X (gfx1250) — compile-verified, matrix operand order checked by reading, not run on hardware
//
#include <hip/hip_runtime.h>

typedef _Float16 v8h  __attribute__((ext_vector_type(8)));
typedef _Float16 v16h __attribute__((ext_vector_type(16)));
typedef float    v8f  __attribute__((ext_vector_type(8)));

// Problem constants (from reference setup_inputs)
static constexpr int   kN = 2048;
static constexpr int   kD = 512;
static constexpr int   kC = 100000;
static constexpr float kS = 30.0f;
static constexpr float kM = 0.5f;

// GEMM tiling
static constexpr int KC      = 32;            // K per WMMA step
static constexpr int XROWS   = 64;            // N rows per workgroup
static constexpr int WROWS   = 256;           // C rows per workgroup
static constexpr int LSTRIDE = 40;            // halves per LDS row (80 B, padded vs 64 B -> bank-conflict free)
static constexpr int XSZ     = XROWS * LSTRIDE;   // 2560 halves
static constexpr int WSZ     = WROWS * LSTRIDE;   // 10240 halves
static constexpr int BUFSZ   = XSZ + WSZ;         // one double-buffer slot (halves)

// LDS offset of a pointer into a __shared__ array (AS3 pointer value == LDS byte offset)
typedef __attribute__((address_space(3))) const void* lds_cptr_t;
#define LDS_BYTE_OFF(p) ((unsigned)(unsigned long long)(lds_cptr_t)(p))

// ---------------------------------------------------------------------------
// Phase 1: row-wise L1 normalize fp32 -> f16 (norm folded into the data)
// one 256-thread block per row, D = 512 (2 elements / thread)
// ---------------------------------------------------------------------------
__global__ __launch_bounds__(256) void l1norm_f16_kernel(const float* __restrict__ src,
                                                         _Float16* __restrict__ dst) {
    const int row = blockIdx.x;
    const int t   = threadIdx.x;
    const size_t base = (size_t)row * kD;
    const float a = src[base + t];
    const float b = src[base + t + 256];

    __shared__ float red[256];
    red[t] = fabsf(a) + fabsf(b);
    __syncthreads();
    #pragma unroll
    for (int s = 128; s > 0; s >>= 1) {
        if (t < s) red[t] += red[t + s];
        __syncthreads();
    }
    const float inv = 1.0f / fmaxf(red[0], 1e-12f);
    dst[base + t]       = (_Float16)(a * inv);
    dst[base + t + 256] = (_Float16)(b * inv);
}

// ---------------------------------------------------------------------------
// Phase 2: cos = xh @ wh^T, scaled by S.  Wave tile 32x64 (8 accumulators),
// workgroup tile 64x256, K double-buffered through LDS in chunks of 32 using
// async global->LDS DMA (ASYNCcnt path).
// ---------------------------------------------------------------------------
__global__ __launch_bounds__(256) void arcface_gemm_kernel(const _Float16* __restrict__ xh,
                                                           const _Float16* __restrict__ wh,
                                                           float* __restrict__ out) {
    __shared__ _Float16 lds[2 * BUFSZ];   // 51200 B

    const int tid    = threadIdx.x;
    const int lane   = tid & 31;
    const int wave   = tid >> 5;
    const int wave_n = wave >> 2;          // 0..1
    const int wave_c = wave & 3;           // 0..3

    const int       n_blk = blockIdx.x * XROWS;
    const long long c_blk = (long long)blockIdx.y * WROWS;

    // staging assignments: X -> 64 rows x 64 B, 16 B/thread; W -> 256 rows x 64 B, 64 B/thread
    const int xrow = tid >> 2;
    const int xcol = (tid & 3) * 8;        // halves (16 B chunks)
    long long wrow = c_blk + tid;
    if (wrow >= kC) wrow = kC - 1;         // clamp; clamped cols are never stored

    const _Float16* xg_base = xh + (size_t)(n_blk + xrow) * kD + xcol;
    const _Float16* wg_base = wh + (size_t)wrow * kD;
    const unsigned  x_lds0  = LDS_BYTE_OFF(&lds[xrow * LSTRIDE + xcol]);
    const unsigned  w_lds0  = LDS_BYTE_OFF(&lds[XSZ + tid * LSTRIDE]);

    // Issue async DMA of one K-chunk (global -> LDS, no VGPR round trip).
    // NOTE (ISA 10.x): IOFFSET is added to BOTH the global and LDS addresses,
    // which matches the contiguous 64 B row segment inside the 80 B LDS row.
    auto issue_chunk = [&](int k0, int b) {
        const _Float16* xg = xg_base + k0;
        const _Float16* wg = wg_base + k0;
        const unsigned  xo = x_lds0 + (unsigned)(b * BUFSZ * 2);
        const unsigned  wo = w_lds0 + (unsigned)(b * BUFSZ * 2);
        asm volatile("global_load_async_to_lds_b128 %0, %1, off"
                     :: "v"(xo), "v"(xg) : "memory");
        asm volatile("global_load_async_to_lds_b128 %0, %1, off"
                     :: "v"(wo), "v"(wg) : "memory");
        asm volatile("global_load_async_to_lds_b128 %0, %1, off offset:16"
                     :: "v"(wo), "v"(wg) : "memory");
        asm volatile("global_load_async_to_lds_b128 %0, %1, off offset:32"
                     :: "v"(wo), "v"(wg) : "memory");
        asm volatile("global_load_async_to_lds_b128 %0, %1, off offset:48"
                     :: "v"(wo), "v"(wg) : "memory");
    };
    auto wait_async = [&]() {
        asm volatile("s_wait_asynccnt 0x0" ::: "memory");
    };

    union Frag { v16h v; v8h h[2]; };
    const int mrow = lane & 15;            // M (A) / N (B) index within 16-tile
    const int koff = (lane >> 4) * 8;      // K chunk offset in halves (ISA 7.12.2 16-bit A/B layout)

    v8f acc[2][4] = {};

    issue_chunk(0, 0);
    wait_async();
    __syncthreads();

    for (int k = 0; k < kD / KC; ++k) {
        const int b = k & 1;
        if (k + 1 < kD / KC) issue_chunk((k + 1) * KC, 1 - b);  // DMA overlaps compute

        const _Float16* base = lds + b * BUFSZ;
        Frag a[2], bf[4];
        #pragma unroll
        for (int i = 0; i < 2; ++i) {
            const _Float16* p = base + (wave_n * 32 + i * 16 + mrow) * LSTRIDE + koff;
            a[i].h[0] = *(const v8h*)(p);        // K = {0..7 | 8..15} of chunk
            a[i].h[1] = *(const v8h*)(p + 16);   // K = {16..23 | 24..31}
        }
        #pragma unroll
        for (int j = 0; j < 4; ++j) {
            const _Float16* p = base + XSZ + (wave_c * 64 + j * 16 + mrow) * LSTRIDE + koff;
            bf[j].h[0] = *(const v8h*)(p);
            bf[j].h[1] = *(const v8h*)(p + 16);
        }
        #pragma unroll
        for (int i = 0; i < 2; ++i)
            #pragma unroll
            for (int j = 0; j < 4; ++j)
                acc[i][j] = __builtin_amdgcn_wmma_f32_16x16x32_f16(
                    false, a[i].v, false, bf[j].v, (short)0, acc[i][j], false, false);

        wait_async();          // this wave's DMA for chunk k+1 has landed
        __syncthreads();       // ... and every other wave's too
    }

    // Epilogue: C/D layout -> lane%16 = N (col), VGPR r = M row (+8 for lanes 16..31)
    const int rown = n_blk + wave_n * 32 + 8 * (lane >> 4);
    #pragma unroll
    for (int i = 0; i < 2; ++i) {
        #pragma unroll
        for (int j = 0; j < 4; ++j) {
            const long long c0 = c_blk + wave_c * 64 + j * 16;
            if (c0 + 16 > kC) continue;           // kC % 16 == 0 -> tiles are all-or-nothing
            const size_t rbase = (size_t)(rown + i * 16) * kC + (size_t)c0 + (lane & 15);
            #pragma unroll
            for (int r = 0; r < 8; ++r)
                out[rbase + (size_t)r * kC] = acc[i][j][r] * kS;
        }
    }
}

// ---------------------------------------------------------------------------
// Phase 3: angular-margin fixup on the label column
// ---------------------------------------------------------------------------
__global__ __launch_bounds__(256) void arcface_margin_kernel(const int* __restrict__ labels,
                                                             float* __restrict__ out) {
    const int n = blockIdx.x * blockDim.x + threadIdx.x;
    if (n >= kN) return;
    const int c = labels[n];
    const size_t idx = (size_t)n * kC + (size_t)c;
    float cosv = out[idx] * (1.0f / kS);
    cosv = fminf(fmaxf(cosv, -1.0f + 1e-7f), 1.0f - 1e-7f);
    out[idx] = cosf(acosf(cosv) + kM) * kS;
}

// ---------------------------------------------------------------------------
extern "C" void kernel_launch(void* const* d_in, const int* in_sizes, int n_in,
                              void* d_out, int out_size, void* d_ws, size_t ws_size,
                              hipStream_t stream) {
    const float* emb    = (const float*)d_in[0];   // [2048, 512] f32
    const float* weight = (const float*)d_in[1];   // [100000, 512] f32
    const int*   labels = (const int*)d_in[2];     // [2048] int
    float*       out    = (float*)d_out;           // [2048, 100000] f32

    // scratch: normalized f16 copies (~100 MB total)
    _Float16* xh = (_Float16*)d_ws;                        // 2048*512 halves
    _Float16* wh = xh + (size_t)kN * kD;                   // 100000*512 halves

    l1norm_f16_kernel<<<kN, 256, 0, stream>>>(emb, xh);
    l1norm_f16_kernel<<<kC, 256, 0, stream>>>(weight, wh);

    dim3 grid(kN / XROWS, (kC + WROWS - 1) / WROWS);       // x = N tiles (fastest) -> L2 reuse of W panels
    arcface_gemm_kernel<<<grid, 256, 0, stream>>>(xh, wh, out);

    arcface_margin_kernel<<<(kN + 255) / 256, 256, 0, stream>>>(labels, out);
}
